// MHA_18562848653934
// MI455X (gfx1250) — matrix operand order repaired and checked by reading, not compiled, not run on hardware
//
#include <hip/hip_runtime.h>
#include <hip/hip_bf16.h>
#include <math.h>

// ---------------------------------------------------------------------------
// MHA block for MI455X (gfx1250, wave32, WMMA + TDM + async global->LDS DMA).
// B=2, T=2048, C=2048, H=16, D=128.
// Pipeline:
//   [cvt x,w_qkv,w_proj -> bf16 (weights transposed to [N,K])]
//   [gemm qkv: bf16 x bf16, double-buffered TDM tile staging]
//   [rmsnorm+rope -> qB,kB ([B,H,T,D]); v -> vBT transposed ([B,H,D,T])]
//   [flash attn: double-buffered async-to-LDS K/V tiles, all frags b128]
//   [gemm proj: same all-bf16 TDM GEMM]
// Matrix math: v_wmma_f32_16x16x32_bf16 (f32 accumulate).
// ---------------------------------------------------------------------------

typedef __bf16 bf16_t;
typedef __attribute__((ext_vector_type(16))) __bf16 v16bf;
typedef __attribute__((ext_vector_type(8)))  __bf16 v8bf;
typedef __attribute__((ext_vector_type(8)))  float  v8f;
typedef __attribute__((ext_vector_type(4)))  float  v4f;
typedef __attribute__((ext_vector_type(4)))  unsigned int v4u;
typedef __attribute__((ext_vector_type(8)))  int v8i;
typedef __attribute__((ext_vector_type(4)))  int v4i;

#define B_   2
#define T_   2048
#define C_   2048
#define H_   16
#define D_   128
#define QKV3 (3 * C_)          // 6144
#define EPS_ 1.1920929e-07f
#define LN_ROPE_BASE 13.815510557964274f   // ln(1e6)

// TDM availability / arity detection:
//   amdgpu-toolchain (clang-23) ships the TDM header and uses a 6-arg builtin;
//   ROCm 7.2 (clang-22) has a 5-arg builtin and no header.
#if defined(__has_include)
#if __has_include(<hip/amd_detail/amd_gfx1250_TDM.h>)
#define TDM_6ARG 1
#endif
#endif
#if defined(__has_builtin)
#if __has_builtin(__builtin_amdgcn_tensor_load_to_lds)
#define HAVE_TDM 1
#endif
#endif

__device__ __forceinline__ bf16_t f2bf(float x) { return (bf16_t)x; }

__device__ __forceinline__ v16bf join8(v8bf lo, v8bf hi) {
  union { v16bf v; v8bf h[2]; } u;
  u.h[0] = lo; u.h[1] = hi;
  return u.v;
}

__device__ __forceinline__ v8f wmma_bf16(v16bf a, v16bf b, v8f c) {
  // D = A(16x32 bf16) * B(32x16 bf16) + C(16x16 f32)
  return __builtin_amdgcn_wmma_f32_16x16x32_bf16(
      /*neg_a=*/false, a, /*neg_b=*/false, b,
      /*c_mod=*/(short)0, c, /*reuse_a=*/false, /*reuse_b=*/false);
}

// Async DMA of one 16B chunk: global -> LDS (GV mode, tracked by ASYNCcnt).
__device__ __forceinline__ void async_ld16(void* lds, const void* g) {
  unsigned la = (unsigned)(uintptr_t)lds;
  asm volatile("global_load_async_to_lds_b128 %0, %1, off"
               :: "v"(la), "v"(g) : "memory");
}
__device__ __forceinline__ void wait_async0() {
  asm volatile("s_wait_asynccnt 0" ::: "memory");
}

#if defined(HAVE_TDM)
// Issue one 2D TDM tile load: tile_d1 rows of tile_d0 bf16 elements, row
// stride stride_d0 elements in global, packed contiguously into LDS.
// D# bitfields per CDNA5 ISA 8.3/8.4 (data_size=1 -> 2 bytes, type=2).
__device__ __forceinline__ void tdm_load_2d(unsigned lds_off, const void* gaddr,
                                            unsigned tile_d0, unsigned tile_d1,
                                            unsigned tensor_d0, unsigned tensor_d1,
                                            unsigned stride_d0) {
  unsigned long long ga = (unsigned long long)(uintptr_t)gaddr;
  v4u g0;
  g0[0] = 1u;                                                  // count=1
  g0[1] = lds_off;                                             // lds_addr (B)
  g0[2] = (unsigned)ga;                                        // addr[31:0]
  g0[3] = (unsigned)((ga >> 32) & 0x01FFFFFFu) | (2u << 30);   // addr[56:32]|type=2
  v8i g1;
  g1[0] = (int)(1u << 16);                                     // data_size=2B
  g1[1] = (int)((tensor_d0 & 0xFFFFu) << 16);                  // tdim0[15:0]
  g1[2] = (int)((tensor_d0 >> 16) | ((tensor_d1 & 0xFFFFu) << 16));
  g1[3] = (int)((tensor_d1 >> 16) | (tile_d0 << 16));          // tile_dim0
  g1[4] = (int)(tile_d1 & 0xFFFFu);                            // tile_dim1 (dim2=0)
  g1[5] = (int)stride_d0;                                      // dim0_stride lo
  g1[6] = 0;                                                   // stride hi / dim1_stride
  g1[7] = 0;
  v4i z = {0, 0, 0, 0};
#if defined(TDM_6ARG)
  v8i z8 = {0, 0, 0, 0, 0, 0, 0, 0};
  __builtin_amdgcn_tensor_load_to_lds(g0, g1, z, z, z8, 0);
#else
  __builtin_amdgcn_tensor_load_to_lds(g0, g1, z, z, 0);
#endif
}
__device__ __forceinline__ void wait_tensor0() {
  __builtin_amdgcn_s_wait_tensorcnt(0);
}
#endif

// ---------------------------------------------------------------------------
// Conversion prepass.
// ---------------------------------------------------------------------------
__global__ __launch_bounds__(256, 1)
void cvt_copy_kernel(const float* __restrict__ src, bf16_t* __restrict__ dst) {
  const size_t i = ((size_t)blockIdx.x * 256 + threadIdx.x) * 8;
  v4f a = *(const v4f*)(src + i);
  v4f b = *(const v4f*)(src + i + 4);
  union { v8bf v; bf16_t e[8]; } u;
#pragma unroll
  for (int j = 0; j < 4; ++j) { u.e[j] = f2bf(a[j]); u.e[4 + j] = f2bf(b[j]); }
  *(v8bf*)(dst + i) = u.v;
}

// dst[n*K + k] = bf16(src[k*N + n]) : weight transpose so GEMM B tiles are
// row-contiguous in k (DMA-able as 2D tiles, [n][k] in LDS).
__global__ __launch_bounds__(256, 1)
void cvt_transpose_kernel(const float* __restrict__ src, bf16_t* __restrict__ dst,
                          int K, int N) {
  const int i = blockIdx.x * 256 + threadIdx.x;
  const int kc = K >> 3;
  const int n = i / kc, k0 = (i % kc) * 8;
  union { v8bf v; bf16_t e[8]; } u;
#pragma unroll
  for (int j = 0; j < 8; ++j) u.e[j] = f2bf(src[(size_t)(k0 + j) * N + n]);
  if (k0 + 8 < K) __builtin_prefetch(&src[(size_t)(k0 + 8) * N + n], 0, 1);
  *(v8bf*)(dst + (size_t)n * K + k0) = u.v;
}

// ---------------------------------------------------------------------------
// All-bf16 GEMM: C[M,N] = A[M,K] @ Bt[N,K]^T, C fp32.
// Block tile 128x64, K-step 32, 8 waves; wave w -> rows 16w..16w+15, 64 cols.
// Tile staging via TDM (2 descriptors per K-step, issued by wave 0, tracked
// by TENSORcnt), double buffered: tile k+1 in flight while tile k feeds the
// WMMAs. Falls back to per-chunk async DMA if the TDM builtin is absent.
// ---------------------------------------------------------------------------
__global__ __launch_bounds__(256, 1)
void gemm_bf16_wmma(const bf16_t* __restrict__ A, const bf16_t* __restrict__ Bt,
                    float* __restrict__ Cout, int M, int N, int K) {
  __shared__ alignas(16) bf16_t As[2][128 * 32];
  __shared__ alignas(16) bf16_t Bs[2][64 * 32];   // [n][k]
  const int tid = threadIdx.x;
  const int wv  = tid >> 5;
  const int ln  = tid & 31;
  const int m0  = blockIdx.y * 128;
  const int n0  = blockIdx.x * 64;

#if defined(HAVE_TDM)
  auto stage = [&](int k0, int bufid) {
    if (tid < 32) {   // one wave issues both tile descriptors
      tdm_load_2d((unsigned)(uintptr_t)&As[bufid][0],
                  &A[(size_t)m0 * K + k0], 32, 128,
                  (unsigned)K, (unsigned)M, (unsigned)K);
      tdm_load_2d((unsigned)(uintptr_t)&Bs[bufid][0],
                  &Bt[(size_t)n0 * K + k0], 32, 64,
                  (unsigned)K, (unsigned)N, (unsigned)K);
    }
  };
  auto stage_wait = [&]() { if (tid < 32) wait_tensor0(); };
#else
  auto stage = [&](int k0, int bufid) {
#pragma unroll
    for (int i = 0; i < 3; ++i) {
      int id = i * 256 + tid;     // 0..767
      if (id < 512) {             // A tile
        int m = id >> 2, ch = id & 3;
        async_ld16(&As[bufid][m * 32 + ch * 8],
                   &A[(size_t)(m0 + m) * K + k0 + ch * 8]);
      } else {                    // B tile
        int rem = id - 512;
        int n = rem >> 2, ch = rem & 3;
        async_ld16(&Bs[bufid][n * 32 + ch * 8],
                   &Bt[(size_t)(n0 + n) * K + k0 + ch * 8]);
      }
    }
  };
  auto stage_wait = [&]() { wait_async0(); };
#endif

  v8f acc[4] = {};
  const int nK = K >> 5;

  stage(0, 0);
  stage_wait();
  __syncthreads();

  for (int kt = 0; kt < nK; ++kt) {
    const int buf = kt & 1;
    if (kt + 1 < nK) stage((kt + 1) << 5, buf ^ 1);

    // A fragment: lane<16 -> row=l, K={0..7,16..23}; lane>=16 -> +8.
    const int row = ln & 15;
    const int khA = (ln >> 4) * 8;
    const bf16_t* ap = &As[buf][(wv * 16 + row) * 32 + khA];
    v16bf afrag = join8(*(const v8bf*)ap, *(const v8bf*)(ap + 16));

    // B fragment: lane holds column N=l&15, K={0..7,16..23}; lane>=16 -> +8.
    const int khB = (ln >> 4) * 8;
#pragma unroll
    for (int ns = 0; ns < 4; ++ns) {
      const bf16_t* bp = &Bs[buf][(ns * 16 + row) * 32 + khB];
      v16bf bfrag = join8(*(const v8bf*)bp, *(const v8bf*)(bp + 16));
      acc[ns] = wmma_bf16(afrag, bfrag, acc[ns]);
    }

    stage_wait();
    __syncthreads();
  }

  // C layout: VGPR r of lane l -> (row = r + 8*(l>>4), col = l&15).
  const int colb = ln & 15;
  const int rofs = (ln >> 4) * 8;
#pragma unroll
  for (int ns = 0; ns < 4; ++ns)
#pragma unroll
    for (int r = 0; r < 8; ++r)
      Cout[(size_t)(m0 + wv * 16 + r + rofs) * N + (n0 + ns * 16 + colb)] =
          acc[ns][r];
}

// ---------------------------------------------------------------------------
// RMSNorm + RoPE on q,k; v converted to bf16 TRANSPOSED ([B,H,D,T]) so the
// attention kernel's V tiles land in LDS as [d][key] (contiguous fragments).
// One wave per (b,h,t); lane owns d = 4*lane..4*lane+3; rotate_half partner
// lives in lane^16 (wave32).
// ---------------------------------------------------------------------------
__global__ __launch_bounds__(256, 1)
void normrope_kernel(const float* __restrict__ qkv, const float* __restrict__ qw,
                     const float* __restrict__ kw, bf16_t* __restrict__ qB,
                     bf16_t* __restrict__ kB, bf16_t* __restrict__ vBT) {
  const int wg = blockIdx.x * 8 + (threadIdx.x >> 5);
  const int ln = threadIdx.x & 31;
  const int h = wg & (H_ - 1);
  const int t = (wg >> 4) & (T_ - 1);
  const int b = wg >> 15;

  const float* base = qkv + (size_t)(b * T_ + t) * QKV3 + h * D_;
  v4f qv = *(const v4f*)(base + ln * 4);
  v4f kv = *(const v4f*)(base + C_ + ln * 4);
  v4f vv = *(const v4f*)(base + 2 * C_ + ln * 4);

  float sq = 0.f, sk = 0.f;
#pragma unroll
  for (int i = 0; i < 4; ++i) { sq += qv[i] * qv[i]; sk += kv[i] * kv[i]; }
#pragma unroll
  for (int off = 16; off >= 1; off >>= 1) {
    sq += __shfl_xor(sq, off, 32);
    sk += __shfl_xor(sk, off, 32);
  }
  const float rq = rsqrtf(sq * (1.f / D_) + EPS_);
  const float rk = rsqrtf(sk * (1.f / D_) + EPS_);

  float qn[4], kn[4];
#pragma unroll
  for (int i = 0; i < 4; ++i) {
    int d = ln * 4 + i;
    qn[i] = qv[i] * rq * qw[d];
    kn[i] = kv[i] * rk * kw[d];
  }
  const size_t bh = (size_t)(b * H_ + h);
  const size_t obase = (bh * T_ + t) * D_;
#pragma unroll
  for (int i = 0; i < 4; ++i) {
    int d = ln * 4 + i;
    float qp = __shfl_xor(qn[i], 16, 32);   // element d^64
    float kp = __shfl_xor(kn[i], 16, 32);
    int   fi = d & 63;
    float invf = __expf(-(float)(2 * fi) * (1.f / D_) * LN_ROPE_BASE);
    float ang = (float)t * invf;
    float sn, cs;
    __sincosf(ang, &sn, &cs);
    float rotq = (d < 64) ? -qp : qp;
    float rotk = (d < 64) ? -kp : kp;
    qB[obase + d] = f2bf(qn[i] * cs + rotq * sn);
    kB[obase + d] = f2bf(kn[i] * cs + rotk * sn);
    vBT[(bh * D_ + d) * T_ + t] = f2bf(vv[i]);   // transposed
  }
}

// ---------------------------------------------------------------------------
// Causal flash attention, double-buffered async K/V staging (ASYNCcnt).
// Block = (qblk,h,b) = 8 waves x 16 q rows; all waves sweep the SAME
// nkb = 4*qblk+4 key blocks so barriers are uniform. Per 32-key block:
// 8 WMMAs for S = q@k^T, online softmax (16-lane shuffles), P transposed
// through per-wave LDS (s_wait_dscnt), 8 WMMAs for y += P@V. All fragment
// reads are contiguous ds_load_b128 pairs (V tile is [d][key] in LDS).
// ---------------------------------------------------------------------------
__global__ __launch_bounds__(256, 1)
void attn_kernel(const bf16_t* __restrict__ qB, const bf16_t* __restrict__ kB,
                 const bf16_t* __restrict__ vBT, bf16_t* __restrict__ yB) {
  __shared__ alignas(16) bf16_t Ks[2][32 * D_];    // [key][d]
  __shared__ alignas(16) bf16_t Vs[2][D_ * 32];    // [d][key]
  __shared__ alignas(16) bf16_t Plds[8 * 16 * 32];
  const int tid = threadIdx.x;
  const int wv = tid >> 5;
  const int ln = tid & 31;
  const int qblk = blockIdx.x, h = blockIdx.y, b = blockIdx.z;
  const int qr0 = qblk * 128 + wv * 16;
  const size_t bh = (size_t)(b * H_ + h);
  const bf16_t* qbase = qB + bh * T_ * D_;
  bf16_t* pl = &Plds[wv * 512];

  const int colb = ln & 15;
  const int rofs = (ln >> 4) * 8;

  // Stage K tile (32x128) + V tile (128x32, transposed source) into buffer:
  // 1024 x 16B async chunks over 256 threads (4 each).
  auto stage = [&](int kbt, int bufid) {
#pragma unroll
    for (int i = 0; i < 4; ++i) {
      int id = i * 256 + tid;          // 0..1023
      if (id < 512) {                  // K: row=key, 16 chunks of d
        int row = id >> 4, ch = id & 15;
        async_ld16(&Ks[bufid][row * D_ + ch * 8],
                   &kB[(bh * T_ + (size_t)kbt * 32 + row) * D_ + ch * 8]);
      } else {                         // V: row=d, 4 chunks of key
        int rem = id - 512;
        int d = rem >> 2, ch = rem & 3;
        async_ld16(&Vs[bufid][d * 32 + ch * 8],
                   &vBT[(bh * D_ + d) * T_ + (size_t)kbt * 32 + ch * 8]);
      }
    }
  };

  // Resident Q fragments (16 rows x 128 d = 4 chunks of K=32).
  v16bf qf[4];
  {
    const bf16_t* qp = qbase + (size_t)(qr0 + colb) * D_;
    const int kh = (ln >> 4) * 8;
#pragma unroll
    for (int c = 0; c < 4; ++c) {
      const bf16_t* p = qp + c * 32 + kh;
      qf[c] = join8(*(const v8bf*)p, *(const v8bf*)(p + 16));
    }
  }

  v8f yacc[8] = {};
  float mrow[8], lrow[8];
#pragma unroll
  for (int r = 0; r < 8; ++r) { mrow[r] = -INFINITY; lrow[r] = 0.f; }

  const float scale = 1.0f / (float)D_;
  const int nkb = qblk * 4 + 4;   // uniform over the block (causal upper bound)

  stage(0, 0);
  wait_async0();
  __syncthreads();

  for (int kb = 0; kb < nkb; ++kb) {
    const int buf = kb & 1;
    if (kb + 1 < nkb) stage(kb + 1, buf ^ 1);   // prefetch next tile

    if (kb * 32 <= qr0 + 15) {   // any unmasked key for this wave's rows?
      const int key0 = kb * 32 + colb;

      // ---- S = q @ k^T for 32 keys: two 16x16 tiles ----
      v8f s0 = {}, s1 = {};
      {
        const int kh = (ln >> 4) * 8;
        const bf16_t* kp0 = &Ks[buf][colb * D_ + kh];
        const bf16_t* kp1 = &Ks[buf][(colb + 16) * D_ + kh];
#pragma unroll
        for (int c = 0; c < 4; ++c) {
          v16bf bk0 = join8(*(const v8bf*)(kp0 + c * 32),
                            *(const v8bf*)(kp0 + c * 32 + 16));
          v16bf bk1 = join8(*(const v8bf*)(kp1 + c * 32),
                            *(const v8bf*)(kp1 + c * 32 + 16));
          s0 = wmma_bf16(qf[c], bk0, s0);
          s1 = wmma_bf16(qf[c], bk1, s1);
        }
      }

      // ---- online softmax (rows live in VGPR r, N across 16 lanes) ----
#pragma unroll
      for (int r = 0; r < 8; ++r) {
        const int rowg = qr0 + r + rofs;
        float v0 = s0[r] * scale; if (key0 > rowg)      v0 = -3.0e38f;
        float v1 = s1[r] * scale; if (key0 + 16 > rowg) v1 = -3.0e38f;
        float tm = fmaxf(v0, v1);
#pragma unroll
        for (int off = 8; off >= 1; off >>= 1)
          tm = fmaxf(tm, __shfl_xor(tm, off, 16));
        const float mnew = fmaxf(mrow[r], tm);
        const float corr = __expf(mrow[r] - mnew);
        const float p0 = __expf(v0 - mnew);
        const float p1 = __expf(v1 - mnew);
        float ps = p0 + p1;
#pragma unroll
        for (int off = 8; off >= 1; off >>= 1)
          ps += __shfl_xor(ps, off, 16);
        lrow[r] = lrow[r] * corr + ps;
        mrow[r] = mnew;
#pragma unroll
        for (int n = 0; n < 8; ++n) yacc[n][r] *= corr;
        const int prow = r + rofs;
        pl[prow * 32 + colb]      = f2bf(p0);
        pl[prow * 32 + 16 + colb] = f2bf(p1);
      }

      // C-layout -> A-layout transpose of P through LDS (cross-lane).
      asm volatile("s_wait_dscnt 0" ::: "memory");
      v16bf pf;
      {
        const bf16_t* pp = pl + (ln & 15) * 32 + (ln >> 4) * 8;
        pf = join8(*(const v8bf*)pp, *(const v8bf*)(pp + 16));
      }

      // ---- y += P @ V; V frags contiguous from [d][key] LDS tile ----
      {
        const int kh16 = (ln >> 4) * 8;
#pragma unroll
        for (int n = 0; n < 8; ++n) {
          const bf16_t* vp = &Vs[buf][(n * 16 + colb) * 32 + kh16];
          v16bf bv = join8(*(const v8bf*)vp, *(const v8bf*)(vp + 16));
          yacc[n] = wmma_bf16(pf, bv, yacc[n]);
        }
      }
    }

    wait_async0();     // next tile landed
    __syncthreads();   // everyone done with `buf`
  }

  // Normalize and scatter y back to [B,T,C] bf16 for the projection GEMM.
  bf16_t* yb = yB + (size_t)b * T_ * C_ + h * D_;
#pragma unroll
  for (int r = 0; r < 8; ++r) {
    const float inv = 1.0f / lrow[r];
    const int trow = qr0 + r + rofs;
#pragma unroll
    for (int n = 0; n < 8; ++n)
      yb[(size_t)trow * C_ + n * 16 + colb] = f2bf(yacc[n][r] * inv);
  }
}

// ---------------------------------------------------------------------------
// Launch. Workspace layout (bytes), with region reuse (stream-ordered):
//   [0, 100663296)           qkvF fp32 [B*T, 6144]
//   A = 100663296:
//     phase 1 (cvt+gemm1):   xB bf16 [4096,2048] @A, wqkvT bf16 @A+16.8MB
//     phase 2 (normrope..):  qB @A, kB @A+16.8MB, vBT @A+33.5MB, yB @A+50.3MB
//   WP = A + 67108864:       wprojT bf16 [2048,2048] (8.4MB)  -> total ~176MB
// ---------------------------------------------------------------------------
extern "C" void kernel_launch(void* const* d_in, const int* in_sizes, int n_in,
                              void* d_out, int out_size, void* d_ws, size_t ws_size,
                              hipStream_t stream) {
  (void)in_sizes; (void)n_in; (void)out_size; (void)ws_size;
  const float* x      = (const float*)d_in[0];
  const float* w_qkv  = (const float*)d_in[1];
  const float* w_proj = (const float*)d_in[2];
  const float* qnw    = (const float*)d_in[3];
  const float* knw    = (const float*)d_in[4];
  float* out = (float*)d_out;

  char* ws = (char*)d_ws;
  float*  qkvF  = (float*)ws;
  char*   regA  = ws + 100663296u;
  bf16_t* xB    = (bf16_t*)regA;                      // phase 1
  bf16_t* wqkvT = (bf16_t*)(regA + 16777216u);        // phase 1
  bf16_t* qB    = (bf16_t*)regA;                      // phase 2 (reuse)
  bf16_t* kB    = (bf16_t*)(regA + 1u * 16777216u);
  bf16_t* vBT   = (bf16_t*)(regA + 2u * 16777216u);
  bf16_t* yB    = (bf16_t*)(regA + 3u * 16777216u);
  bf16_t* wprojT = (bf16_t*)(regA + 4u * 16777216u);  // persistent

  dim3 blk(256);

  // Prepass: bf16 conversion (+ weight transposes to [N,K]).
  cvt_copy_kernel<<<(B_ * T_ * C_) / (8 * 256), blk, 0, stream>>>(x, xB);
  cvt_transpose_kernel<<<(C_ * QKV3) / (8 * 256), blk, 0, stream>>>(w_qkv, wqkvT, C_, QKV3);
  cvt_transpose_kernel<<<(C_ * C_) / (8 * 256), blk, 0, stream>>>(w_proj, wprojT, C_, C_);

  // QKV GEMM.
  dim3 g1(QKV3 / 64, (B_ * T_) / 128);
  gemm_bf16_wmma<<<g1, blk, 0, stream>>>(xB, wqkvT, qkvF, B_ * T_, QKV3, C_);

  // Norm + RoPE (+ transposed V). Overwrites xB/wqkvT region (gemm1 done).
  normrope_kernel<<<(B_ * H_ * T_) / 8, blk, 0, stream>>>(qkvF, qnw, knw, qB, kB, vBT);

  // Flash attention.
  dim3 g3(T_ / 128, H_, B_);
  attn_kernel<<<g3, blk, 0, stream>>>(qB, kB, vBT, yB);

  // Output projection.
  dim3 g4(C_ / 64, (B_ * T_) / 128);
  gemm_bf16_wmma<<<g4, blk, 0, stream>>>(yB, wprojT, out, B_ * T_, C_, C_);
}
